// HGClayer_14499809591691
// MI455X (gfx1250) — matrix operand order repaired and checked by reading, compiled, not run on hardware
//
#include <hip/hip_runtime.h>
#include <hip/hip_bf16.h>

// Problem dims (match reference)
#define NN 16384
#define EE 8192
#define DD 256

typedef __attribute__((ext_vector_type(16))) __bf16 v16bf;
typedef __attribute__((ext_vector_type(8)))  float  v8f;

__device__ __forceinline__ unsigned f32bits(float f) {
    union { float f; unsigned u; } c; c.f = f;
    return c.u;
}

// float -> bf16 bits (truncate; H entries 0/1 are exact, activations lose <=1ulp)
__device__ __forceinline__ unsigned f2bfbits(float f) {
    return f32bits(f) >> 16;
}

// pack two floats into one dword of bf16 pair (lo = first): single v_perm_b32
__device__ __forceinline__ unsigned pack2bf(float lo, float hi) {
    return __builtin_amdgcn_perm(f32bits(hi), f32bits(lo), 0x07060302u);
}

// Load a 16-half WMMA fragment from LDS (pre-swizzled): two b128 LDS reads
__device__ __forceinline__ v16bf ldfrag(const unsigned short* p) {
    union { uint4 q[2]; v16bf v; } u;
    const uint4* q = (const uint4*)p;
    u.q[0] = q[0];
    u.q[1] = q[1];
    return u.v;
}

// A-matrix (16x32 bf16) element (m,k) -> (lane, half) per CDNA5 ISA layout
__device__ __forceinline__ void amap(int m, int k, int& lane, int& h) {
    lane = m + (((k >> 3) & 1) << 4);
    int vg = ((k >> 4) << 2) | ((k & 7) >> 1);
    h = (vg << 1) | (k & 1);
}

// ---------------------------------------------------------------------------
// Wave-per-row matvec: out[row] = dot(A[row, 0:256], v[0:256])
// ---------------------------------------------------------------------------
__global__ __launch_bounds__(256) void dot_rows(const float* __restrict__ A,
                                                const float* __restrict__ v,
                                                float* __restrict__ out,
                                                int rows) {
    int t = threadIdx.x;
    int lane = t & 31;
    int w = t >> 5;
    int row = blockIdx.x * 8 + w;
    if (row >= rows) return;
    float s = 0.f;
    const float* ap = A + (size_t)row * DD;
#pragma unroll
    for (int j = 0; j < 8; ++j)
        s += ap[lane + 32 * j] * v[lane + 32 * j];
#pragma unroll
    for (int off = 16; off > 0; off >>= 1)
        s += __shfl_down(s, off, 32);
    if (lane == 0) out[row] = s;
}

// Stage a K=32 x 256 B tile (rows ks..ks+31 of a [*,256] f32 matrix) into
// fragment-order bf16 LDS: per thread, 2 lane-rows x 2 ds_store_b128.
__device__ __forceinline__ void stage_B(const float* __restrict__ B, int ks,
                                        int t, unsigned short (*Bs)[32][16]) {
#pragma unroll
    for (int r = 0; r < 2; ++r) {
        int ln = (t & 15) + 16 * r;
        unsigned pk[8];
#pragma unroll
        for (int j = 0; j < 8; ++j) {
            float v0 = B[(size_t)(ks + r * 8 + 2 * j + ((j >> 2) << 3)) * DD + t];
            float v1 = B[(size_t)(ks + r * 8 + 2 * j + ((j >> 2) << 3) + 1) * DD + t];
            pk[j] = pack2bf(v0, v1);
        }
        uint4* dst = (uint4*)(&Bs[t >> 4][ln][0]);
        uint4 q0; q0.x = pk[0]; q0.y = pk[1]; q0.z = pk[2]; q0.w = pk[3];
        uint4 q1; q1.x = pk[4]; q1.y = pk[5]; q1.z = pk[6]; q1.w = pk[7];
        dst[0] = q0;
        dst[1] = q1;
    }
}

// ---------------------------------------------------------------------------
// Generic bf16-WMMA GEMM: C[M,256] = A[M,256] @ B[256,256], all f32 row-major.
// Block = 8 waves, tile = 32 rows x 256 cols, K loop of 8 steps of 32.
// ---------------------------------------------------------------------------
__global__ __launch_bounds__(256) void gemm256_bf16(const float* __restrict__ A,
                                                    const float* __restrict__ B,
                                                    float* __restrict__ C) {
    __shared__ alignas(16) unsigned short As[2][32][16];   // 2 KB
    __shared__ alignas(16) unsigned short Bs[16][32][16];  // 16 KB

    const int t = threadIdx.x;
    const int lane = t & 31;
    const int w = t >> 5;
    const int wm = w >> 2;   // 0..1 : which 16-row A tile
    const int wn = w & 3;    // 0..3 : which 64-col band
    const int row0 = blockIdx.x * 32;

    v8f acc[4];
    const v8f vzero = {0.f,0.f,0.f,0.f,0.f,0.f,0.f,0.f};
#pragma unroll
    for (int j = 0; j < 4; ++j) acc[j] = vzero;

    // A staging coords: thread t handles (mt, m, k..k+3) -> one lane, h0..h0+3
    const int a_mt = t >> 7;
    const int a_m  = (t >> 3) & 15;
    const int a_k  = (t & 7) * 4;
    int a_ln, a_h0;
    amap(a_m, a_k, a_ln, a_h0);

    for (int ks = 0; ks < DD; ks += 32) {
        // ---- stage A tile (32 x 32 f32 -> bf16 frags), one b64 store ----
        {
            float4 av = *(const float4*)(A + (size_t)(row0 + a_mt * 16 + a_m) * DD + ks + a_k);
            uint2 p;
            p.x = pack2bf(av.x, av.y);
            p.y = pack2bf(av.z, av.w);
            *(uint2*)(&As[a_mt][a_ln][a_h0]) = p;
        }
        // ---- stage B tile (32 x 256), packed b128 stores ----
        stage_B(B, ks, t, Bs);
        __syncthreads();
        v16bf af = ldfrag(As[wm][lane]);
#pragma unroll
        for (int j = 0; j < 4; ++j) {
            v16bf bf = ldfrag(Bs[wn * 4 + j][lane]);
            acc[j] = __builtin_amdgcn_wmma_f32_16x16x32_bf16(
                false, af, false, bf, (short)0, acc[j], false, false);
        }
        __syncthreads();
    }
    // ---- store C ----
#pragma unroll
    for (int j = 0; j < 4; ++j) {
        int col = wn * 64 + j * 16 + (lane & 15);
#pragma unroll
        for (int v = 0; v < 8; ++v) {
            int rl = wm * 16 + v + 8 * (lane >> 4);
            C[(size_t)(row0 + rl) * DD + col] = acc[j][v];
        }
    }
}

// ---------------------------------------------------------------------------
// e = (H^T @ xw) / edge_deg     (M=E tile of 32, N=256 full, K=N nodes=16384)
// edge_deg fused: accumulated from the same staged H values.
// ---------------------------------------------------------------------------
__global__ __launch_bounds__(256) void hT_gemm(const float* __restrict__ H,
                                               const float* __restrict__ xw,
                                               float* __restrict__ e_out) {
    __shared__ alignas(16) unsigned short As[2][32][16];
    __shared__ alignas(16) unsigned short Bs[16][32][16];
    __shared__ float degf[32];

    const int t = threadIdx.x;
    const int lane = t & 31;
    const int w = t >> 5;
    const int wm = w >> 2;
    const int wn = w & 3;
    const int e0 = blockIdx.x * 32;

    if (t < 32) degf[t] = 0.f;

    v8f acc[4];
    const v8f vzero = {0.f,0.f,0.f,0.f,0.f,0.f,0.f,0.f};
#pragma unroll
    for (int j = 0; j < 4; ++j) acc[j] = vzero;

    // A[m=e_local, k=node]: H[k, e0+m]. Thread: fixed k, 4 consecutive e
    // (m varies -> 4 different lanes, so scalar b16 stores; keeps the H
    // global read a coalesced float4). H values 0/1 -> truncation exact.
    const int kA  = t >> 3;         // 0..31
    const int mlA = (t << 2) & 31;  // 0,4,...,28
    float degp[4] = {0.f, 0.f, 0.f, 0.f};

    for (int ks = 0; ks < NN; ks += 32) {
        const float* hrow = H + (size_t)(ks + kA) * EE + e0 + mlA;
        float4 hv = *(const float4*)hrow;
        if (ks + 32 < NN)
            __builtin_prefetch(hrow + (size_t)32 * EE, 0, 1);
        float hvals[4] = {hv.x, hv.y, hv.z, hv.w};
#pragma unroll
        for (int i = 0; i < 4; ++i) {
            int ml = mlA + i;
            int ln, h; amap(ml & 15, kA, ln, h);
            As[ml >> 4][ln][h] = (unsigned short)f2bfbits(hvals[i]);
            degp[i] += hvals[i];
        }
        stage_B(xw, ks, t, Bs);
        __syncthreads();
        v16bf af = ldfrag(As[wm][lane]);
#pragma unroll
        for (int j = 0; j < 4; ++j) {
            v16bf bf = ldfrag(Bs[wn * 4 + j][lane]);
            acc[j] = __builtin_amdgcn_wmma_f32_16x16x32_bf16(
                false, af, false, bf, (short)0, acc[j], false, false);
        }
        __syncthreads();
    }
#pragma unroll
    for (int i = 0; i < 4; ++i) atomicAdd(&degf[mlA + i], degp[i]);
    __syncthreads();

#pragma unroll
    for (int j = 0; j < 4; ++j) {
        int col = wn * 64 + j * 16 + (lane & 15);
#pragma unroll
        for (int v = 0; v < 8; ++v) {
            int rl = wm * 16 + v + 8 * (lane >> 4);
            float inv = 1.f / degf[rl];
            e_out[(size_t)(e0 + rl) * DD + col] = acc[j][v] * inv;
        }
    }
}

// ---------------------------------------------------------------------------
// out = leaky( (W @ ew) / rowsum ),  W[n,e] = H[n,e]*exp(f(ea2[e]+xa1[n]))
// f(z) = leaky_relu(tanh(z/8)*8). rowsum fused in-pass (block covers full E).
// ---------------------------------------------------------------------------
__global__ __launch_bounds__(256) void attn_out(const float* __restrict__ H,
                                                const float* __restrict__ ew,
                                                const float* __restrict__ ea2,
                                                const float* __restrict__ xa1,
                                                float* __restrict__ out) {
    __shared__ alignas(16) unsigned short As[2][32][16];
    __shared__ alignas(16) unsigned short Bs[16][32][16];
    __shared__ float rsum[32];

    const int t = threadIdx.x;
    const int lane = t & 31;
    const int w = t >> 5;
    const int wm = w >> 2;
    const int wn = w & 3;
    const int n0 = blockIdx.x * 32;

    if (t < 32) rsum[t] = 0.f;

    v8f acc[4];
    const v8f vzero = {0.f,0.f,0.f,0.f,0.f,0.f,0.f,0.f};
#pragma unroll
    for (int j = 0; j < 4; ++j) acc[j] = vzero;

    // W[m=n_local, k=edge]: thread owns fixed row mA, 4 consecutive k
    // -> same lane, consecutive halves -> one b64 LDS store.
    const int mA = t >> 3;          // 0..31
    const int kA = (t << 2) & 31;   // 0,4,...,28
    int a_ln, a_h0;
    amap(mA & 15, kA, a_ln, a_h0);
    const float xr = xa1[n0 + mA];
    float rsp = 0.f;

    for (int ks = 0; ks < EE; ks += 32) {
        const float* hrow = H + (size_t)(n0 + mA) * EE + ks + kA;
        float4 hv = *(const float4*)hrow;
        float4 ev = *(const float4*)(ea2 + ks + kA);
        if (ks + 32 < EE)
            __builtin_prefetch(hrow + 32, 0, 1);
        float hvals[4] = {hv.x, hv.y, hv.z, hv.w};
        float evals[4] = {ev.x, ev.y, ev.z, ev.w};
        float wvals[4];
#pragma unroll
        for (int i = 0; i < 4; ++i) {
            float wv = 0.f;
            if (hvals[i] != 0.f) {
                float s = evals[i] + xr;
                s = tanhf(s * 0.125f) * 8.f;
                s = (s >= 0.f) ? s : 0.1f * s;
                wv = hvals[i] * __expf(s);
            }
            rsp += wv;
            wvals[i] = wv;
        }
        {
            uint2 p;
            p.x = pack2bf(wvals[0], wvals[1]);
            p.y = pack2bf(wvals[2], wvals[3]);
            *(uint2*)(&As[mA >> 4][a_ln][a_h0]) = p;
        }
        stage_B(ew, ks, t, Bs);
        __syncthreads();
        v16bf af = ldfrag(As[wm][lane]);
#pragma unroll
        for (int j = 0; j < 4; ++j) {
            v16bf bf = ldfrag(Bs[wn * 4 + j][lane]);
            acc[j] = __builtin_amdgcn_wmma_f32_16x16x32_bf16(
                false, af, false, bf, (short)0, acc[j], false, false);
        }
        __syncthreads();
    }
    atomicAdd(&rsum[mA], rsp);
    __syncthreads();

#pragma unroll
    for (int j = 0; j < 4; ++j) {
        int col = wn * 64 + j * 16 + (lane & 15);
#pragma unroll
        for (int v = 0; v < 8; ++v) {
            int rl = wm * 16 + v + 8 * (lane >> 4);
            float val = acc[j][v] / rsum[rl];
            val = (val >= 0.f) ? val : 0.1f * val;
            out[(size_t)(n0 + rl) * DD + col] = val;
        }
    }
}

// ---------------------------------------------------------------------------
// Host-side orchestration
// ---------------------------------------------------------------------------
extern "C" void kernel_launch(void* const* d_in, const int* in_sizes, int n_in,
                              void* d_out, int out_size, void* d_ws, size_t ws_size,
                              hipStream_t stream) {
    (void)in_sizes; (void)n_in; (void)out_size; (void)ws_size;
    const float* x     = (const float*)d_in[0];   // [N, 256]
    const float* H     = (const float*)d_in[1];   // [N, E]
    const float* w_v2e = (const float*)d_in[2];   // [256, 256]
    const float* w_e2v = (const float*)d_in[3];   // [256, 256]
    const float* w_v   = (const float*)d_in[4];   // [256, 256]
    const float* a1    = (const float*)d_in[5];   // [256]
    const float* a2    = (const float*)d_in[6];   // [256]
    float* out = (float*)d_out;                   // [N, 256]

    // workspace layout (floats)
    float* ws   = (float*)d_ws;
    float* xw   = ws;                               // N*256
    float* e_   = xw + (size_t)NN * DD;             // E*256
    float* ew   = e_ + (size_t)EE * DD;             // E*256
    float* wa1  = ew + (size_t)EE * DD;             // 256
    float* wa2  = wa1 + DD;                         // 256
    float* xa1  = wa2 + DD;                         // N
    float* ea2  = xa1 + NN;                         // E

    // tiny matvecs: wa1 = w_v @ a1 ; wa2 = w_e2v @ a2
    dot_rows<<<DD / 8, 256, 0, stream>>>(w_v,   a1, wa1, DD);
    dot_rows<<<DD / 8, 256, 0, stream>>>(w_e2v, a2, wa2, DD);
    // xa1 = x @ wa1
    dot_rows<<<NN / 8, 256, 0, stream>>>(x, wa1, xa1, NN);
    // xw = x @ w_v2e
    gemm256_bf16<<<NN / 32, 256, 0, stream>>>(x, w_v2e, xw);
    // e = (H^T @ xw) / edge_deg   (deg fused)
    hT_gemm<<<EE / 32, 256, 0, stream>>>(H, xw, e_);
    // ea2 = e @ wa2
    dot_rows<<<EE / 8, 256, 0, stream>>>(e_, wa2, ea2, EE);
    // ew = e @ w_e2v
    gemm256_bf16<<<EE / 32, 256, 0, stream>>>(e_, w_e2v, ew);
    // out = leaky( softmax-weighted aggregation )
    attn_out<<<NN / 32, 256, 0, stream>>>(H, ew, ea2, xa1, out);
}
